// GraphDetector_89275190215242
// MI455X (gfx1250) — compile-verified, matrix operand order checked by reading, not run on hardware
//
#include <hip/hip_runtime.h>

typedef __attribute__((ext_vector_type(16))) __bf16 v16bf;
typedef __attribute__((ext_vector_type(8)))  float  v8f;

// K-pair base for 16-bit WMMA A fragments (16x32), wave32 layout:
// VGPR v (0..3): K = 2v,2v+1 ; VGPR v (4..7): K = 16+2(v-4),+1 ; +8 for lanes 16..31.
__device__ __forceinline__ int pairbase(int v, int hi) {
    return (v < 4 ? 2 * v : 16 + 2 * (v - 4)) + (hi ? 8 : 0);
}

// Packed-B layout permutation: within a 32-K block, storage order is
// [k=0..7 | k=16..23 | k=8..15 | k=24..31] so that lane-group g's 16 fragment
// elements are the contiguous 32B chunk at offset g*16 (one v16bf load ->
// two global_load_b128 directly into the WMMA source registers).
__device__ __forceinline__ int kperm(int k) {
    return (k & 7) | (((k >> 4) & 1) << 3) | (((k >> 3) & 1) << 4);
}
// Packed element index for B[k][c] with total 'cols' columns:
//   ((k>>5)*cols + c)*32 + kperm(k&31)

// ---------------------------------------------------------------------------
// Generic batched GEMM: C[b] = act( A[b] @ B (+ A2[b] @ B2) + bias )
// A: f32 (cvt to bf16 in-register). B/B2: bf16 in packed fragment layout.
// One wave computes a 16x64 C tile; 4 waves/WG -> 64x64; grid (M/64, N/64, batch).
// ACT: 0 none, 1 ELU, 2 tanh. Cf: f32 row-major [ldc]. Cb: bf16 packed layout
// (for feeding the next WMMA consumer), stride = Ncols.
// ---------------------------------------------------------------------------
template <int ACT>
__global__ __launch_bounds__(128) void gemm_bf16_k(
    const float* __restrict__ A, long bsA, int lda,
    const float* __restrict__ A2,
    const __bf16* __restrict__ B, const __bf16* __restrict__ B2,
    int Ncols, int K,
    const float* __restrict__ bias,
    float* __restrict__ Cf, __bf16* __restrict__ Cb,
    long bsC, int ldc)
{
    const int b = blockIdx.z;
    A += (long)b * bsA;
    const float* A2b = A2 ? A2 + (long)b * bsA : nullptr;
    const int lane = threadIdx.x & 31;
    const int wave = threadIdx.x >> 5;
    const int li = lane & 15;
    const int hi = lane >> 4;
    const int row0 = blockIdx.x * 64 + wave * 16;
    const int col0 = blockIdx.y * 64;

    v8f acc[4] = {};
    for (int k0 = 0; k0 < K; k0 += 32) {
        const int kb = k0 >> 5;
        v16bf a;
#pragma unroll
        for (int v = 0; v < 8; ++v) {
            int kk = pairbase(v, hi);
            const float* ap = A + (long)(row0 + li) * lda + k0 + kk;
            a[2 * v]     = (__bf16)ap[0];
            a[2 * v + 1] = (__bf16)ap[1];
        }
        const __bf16* bbase = B + ((long)kb * Ncols + col0 + li) * 32 + hi * 16;
#pragma unroll
        for (int nt = 0; nt < 4; ++nt) {
            v16bf bb = *(const v16bf*)(bbase + nt * 16 * 32);
            acc[nt] = __builtin_amdgcn_wmma_f32_16x16x32_bf16(
                false, a, false, bb, (short)0, acc[nt], false, false);
        }
        if (A2b) {
            v16bf a2;
#pragma unroll
            for (int v = 0; v < 8; ++v) {
                int kk = pairbase(v, hi);
                const float* ap = A2b + (long)(row0 + li) * lda + k0 + kk;
                a2[2 * v]     = (__bf16)ap[0];
                a2[2 * v + 1] = (__bf16)ap[1];
            }
            const __bf16* b2base = B2 + ((long)kb * Ncols + col0 + li) * 32 + hi * 16;
#pragma unroll
            for (int nt = 0; nt < 4; ++nt) {
                v16bf bb = *(const v16bf*)(b2base + nt * 16 * 32);
                acc[nt] = __builtin_amdgcn_wmma_f32_16x16x32_bf16(
                    false, a2, false, bb, (short)0, acc[nt], false, false);
            }
        }
    }
#pragma unroll
    for (int nt = 0; nt < 4; ++nt) {
        const int col = col0 + nt * 16 + li;
        float vals[8];
#pragma unroll
        for (int v = 0; v < 8; ++v) {
            float val = acc[nt][v];
            if (bias) val += bias[col];
            if (ACT == 1) val = val > 0.f ? val : expm1f(val);
            else if (ACT == 2) val = tanhf(val);
            vals[v] = val;
        }
        if (Cf) {
#pragma unroll
            for (int v = 0; v < 8; ++v) {
                int row = row0 + v + hi * 8;     // C layout: VGPR v -> row v / v+8
                Cf[(long)b * bsC + (long)row * ldc + col] = vals[v];
            }
        }
        if (Cb) {
            // packed layout; rows (v, v+1) are adjacent K values in one lane
#pragma unroll
            for (int v = 0; v < 8; v += 2) {
                int row = row0 + v + hi * 8;
                long idx = (long)b * bsC + (((long)(row >> 5) * Ncols + col) * 32 + kperm(row & 31));
                union { __bf16 h[2]; unsigned int u; } pk;
                pk.h[0] = (__bf16)vals[v];
                pk.h[1] = (__bf16)vals[v + 1];
                *(unsigned int*)(Cb + idx) = pk.u;
            }
        }
    }
}

// ---------------------------------------------------------------------------
// s_src / s_dst: per-(node,head) dot of h[n,h,:] with a_src/a_dst.
// hfeat is in packed fragment layout ([j>>5][col][kperm(j&31)], cols = 4*O).
// ---------------------------------------------------------------------------
__global__ void ssd_k(const __bf16* __restrict__ hfeat,
                      const float* __restrict__ asrc, const float* __restrict__ adst,
                      float* __restrict__ ssrc, float* __restrict__ sdst,
                      int O, long total)
{
    long t = blockIdx.x * (long)blockDim.x + threadIdx.x;
    if (t >= total) return;
    int h = (int)(t & 3);
    long node = t >> 2;              // batch*N + j, batches of 1024 rows (mult of 32)
    const int HO = 4 * O;
    const __bf16* hp = hfeat + ((node >> 5) * (long)HO + h * O) * 32 + kperm((int)(node & 31));
    const float* as = asrc + h * O;
    const float* ad = adst + h * O;
    float s1 = 0.f, s2 = 0.f;
    for (int o = 0; o < O; ++o) {
        float v = (float)hp[(long)o * 32];
        s1 += v * as[o];
        s2 += v * ad[o];
    }
    ssrc[t] = s1;
    sdst[t] = s2;
}

// ---------------------------------------------------------------------------
// Flash-style GAT attention + aggregation.  One WG = 16 target rows, 4 waves =
// 4 heads.  Mask slab (edge[:, i0:i0+16] -> bytes, [j][i]) and s_src staged in
// LDS once per WG; online softmax over j in tiles of 32; P@H via bf16 WMMA
// (f32 accumulate); hfeat B operand in packed layout -> straight b128 loads.
// Heads merged (mean + bias [+ ELU]) through LDS.  NT = O/16.
// ---------------------------------------------------------------------------
template <int NT>
__global__ __launch_bounds__(128) void gat_attn_k(
    const __bf16* __restrict__ hfeat, long hb,
    const float* __restrict__ ssrc, const float* __restrict__ sdst, long sb,
    const int* __restrict__ edge, long eb,
    const float* __restrict__ bias,
    float* __restrict__ out, long ob, int do_elu)
{
    constexpr int O = NT * 16;
    const int N = 1024;
    __shared__ unsigned char maskS[1024 * 16];   // 16 KB, mask[j*16 + i]
    __shared__ float ssrcS[1024 * 4];            // 16 KB, [j][h]
    __shared__ float comb[4][16][O];             // 32 KB (NT=8) / 16 KB (NT=4)

    const int b = blockIdx.y;
    hfeat += (long)b * hb;
    ssrc  += (long)b * sb;
    sdst  += (long)b * sb;
    edge  += (long)b * eb;
    out   += (long)b * ob;

    const int i0 = blockIdx.x * 16;
    const int tid = threadIdx.x;

    for (int idx = tid; idx < N * 16; idx += 128) {
        int ii = idx & 15;
        int j  = idx >> 4;
        int e  = edge[j * N + i0 + ii];
        maskS[idx] = (unsigned char)((e != 0) || (j == i0 + ii));   // adj.T | eye
    }
    for (int idx = tid; idx < N * 4; idx += 128) ssrcS[idx] = ssrc[idx];
    __syncthreads();

    const int lane = tid & 31;
    const int h    = tid >> 5;          // wave == head
    const int li   = lane & 15;
    const int hi   = lane >> 4;
    const int HO   = 4 * O;
    const float sdst_i = sdst[(i0 + li) * 4 + h];

    float m = -1e30f, lsum = 0.f;
    v8f acc[NT] = {};

    for (int jt = 0; jt < N / 32; ++jt) {
        __builtin_prefetch(hfeat + ((long)(jt + 1) * HO + h * O) * 32, 0, 1);
        float p[16];
        float tmax = -1e30f;
#pragma unroll
        for (int v = 0; v < 8; ++v) {
            int kb = pairbase(v, hi);
            int j  = jt * 32 + kb;
#pragma unroll
            for (int e = 0; e < 2; ++e) {
                int jj = j + e;
                float sv  = ssrcS[jj * 4 + h] + sdst_i;
                float lg  = sv > 0.f ? sv : 0.2f * sv;          // LeakyReLU(0.2)
                float val = maskS[jj * 16 + li] ? lg : -1e30f;
                p[2 * v + e] = val;
                tmax = fmaxf(tmax, val);
            }
        }
        tmax = fmaxf(tmax, __shfl_xor(tmax, 16, 32));           // join both K halves
        float mnew = fmaxf(m, tmax);
        float r = (m >= mnew) ? 1.f : __expf(m - mnew);

        float rs = 0.f;
        v16bf afrag;
#pragma unroll
        for (int q = 0; q < 16; ++q) {
            float pe = (p[q] > -1e29f) ? __expf(p[q] - mnew) : 0.f;
            rs += pe;
            afrag[q] = (__bf16)pe;
        }
        rs += __shfl_xor(rs, 16, 32);
        lsum = lsum * r + rs;
        m = mnew;

        float rv[8];
#pragma unroll
        for (int v = 0; v < 8; ++v) rv[v] = __shfl(r, v + (hi << 3), 32);
#pragma unroll
        for (int nt = 0; nt < NT; ++nt)
#pragma unroll
            for (int v = 0; v < 8; ++v) acc[nt][v] *= rv[v];

        const __bf16* bbase = hfeat + ((long)jt * HO + h * O + li) * 32 + hi * 16;
#pragma unroll
        for (int nt = 0; nt < NT; ++nt) {
            v16bf bfrag = *(const v16bf*)(bbase + nt * 16 * 32);
            acc[nt] = __builtin_amdgcn_wmma_f32_16x16x32_bf16(
                false, afrag, false, bfrag, (short)0, acc[nt], false, false);
        }
    }

#pragma unroll
    for (int v = 0; v < 8; ++v) {
        float lrow = __shfl(lsum, v + (hi << 3), 32);
        float inv  = 1.f / lrow;
        int row    = v + hi * 8;
#pragma unroll
        for (int nt = 0; nt < NT; ++nt)
            comb[h][row][nt * 16 + li] = acc[nt][v] * inv;
    }
    __syncthreads();

    for (int idx = tid; idx < 16 * O; idx += 128) {
        int row = idx / O, col = idx % O;
        float s = (comb[0][row][col] + comb[1][row][col] +
                   comb[2][row][col] + comb[3][row][col]) * 0.25f + bias[col];
        if (do_elu) s = s > 0.f ? s : expm1f(s);
        out[(long)(i0 + row) * O + col] = s;
    }
}

// ---------------------------------------------------------------------------
// Strided f32 -> bf16 weight repack into packed fragment layout:
//   dst[((r>>5)*cols + c)*32 + kperm(r&31)] = src[off + r*rstride + c*cstride]
// ---------------------------------------------------------------------------
__global__ void repack_k(const float* __restrict__ src, __bf16* __restrict__ dst,
                         int rows, int cols, int rstride, int cstride, int off)
{
    int idx = blockIdx.x * blockDim.x + threadIdx.x;
    if (idx >= rows * cols) return;
    int r = idx / cols, c = idx % cols;
    long di = ((long)(r >> 5) * cols + c) * 32 + kperm(r & 31);
    dst[di] = (__bf16)src[off + r * rstride + c * cstride];
}

__global__ void copy_k(const float* __restrict__ src, float* __restrict__ dst, int n)
{
    int i = blockIdx.x * blockDim.x + threadIdx.x;
    if (i < n) dst[i] = src[i];
}

// ---------------------------------------------------------------------------
extern "C" void kernel_launch(void* const* d_in, const int* in_sizes, int n_in,
                              void* d_out, int out_size, void* d_ws, size_t ws_size,
                              hipStream_t stream)
{
    const int N = 1024;
    const float* x     = (const float*)d_in[0];
    const int*   edge  = (const int*)d_in[1];
    const float* g1_W  = (const float*)d_in[2];
    const float* g1_as = (const float*)d_in[3];
    const float* g1_ad = (const float*)d_in[4];
    const float* g1_b  = (const float*)d_in[5];
    const float* c1_w  = (const float*)d_in[6];
    const float* c1_b  = (const float*)d_in[7];
    const float* g2_W  = (const float*)d_in[8];
    const float* g2_as = (const float*)d_in[9];
    const float* g2_ad = (const float*)d_in[10];
    const float* g2_b  = (const float*)d_in[11];
    const float* c2_w  = (const float*)d_in[12];
    const float* c2_b  = (const float*)d_in[13];
    const float* r1_W  = (const float*)d_in[14];
    const float* r1_as = (const float*)d_in[15];
    const float* r1_ad = (const float*)d_in[16];
    const float* r1_b  = (const float*)d_in[17];
    const float* r2_W  = (const float*)d_in[18];
    const float* r2_b  = (const float*)d_in[19];
    const float* r3_W  = (const float*)d_in[20];
    const float* r3_b  = (const float*)d_in[21];
    const float* f1_W  = (const float*)d_in[22];
    const float* f1_as = (const float*)d_in[23];
    const float* f1_ad = (const float*)d_in[24];
    const float* f1_b  = (const float*)d_in[25];
    const float* f3_W  = (const float*)d_in[26];
    const float* f3_b  = (const float*)d_in[27];
    (void)in_sizes; (void)n_in; (void)out_size; (void)ws_size;

    char* ws = (char*)d_ws;
    size_t off = 0;
    auto alloc = [&](size_t bytes) -> char* {
        char* p = ws + off;
        off += (bytes + 255) & ~(size_t)255;
        return p;
    };

    // bf16 weights (packed fragment layout)
    __bf16* g1Wb  = (__bf16*)alloc(64 * 512 * 2);
    __bf16* c1w0b = (__bf16*)alloc(128 * 128 * 2);
    __bf16* c1w1b = (__bf16*)alloc(128 * 128 * 2);
    __bf16* g2Wb  = (__bf16*)alloc(128 * 256 * 2);
    __bf16* c2w0b = (__bf16*)alloc(64 * 64 * 2);
    __bf16* c2w1b = (__bf16*)alloc(64 * 64 * 2);
    __bf16* r1Wb  = (__bf16*)alloc(64 * 512 * 2);
    __bf16* r2Wb  = (__bf16*)alloc(128 * 64 * 2);
    __bf16* r3Wb  = (__bf16*)alloc(64 * 64 * 2);
    __bf16* f1Wb  = (__bf16*)alloc(64 * 512 * 2);
    __bf16* f3Wb  = (__bf16*)alloc(64 * 64 * 2);

    // activations
    __bf16* hf1   = (__bf16*)alloc(16L * N * 512 * 2);
    float*  ss1   = (float*)alloc(16L * N * 4 * 4);
    float*  sd1   = (float*)alloc(16L * N * 4 * 4);
    float*  o1    = (float*)alloc(16L * N * 128 * 4);
    float*  o1c   = (float*)alloc(15L * N * 128 * 4);
    __bf16* hf2   = (__bf16*)alloc(15L * N * 256 * 2);
    float*  ss2   = (float*)alloc(15L * N * 4 * 4);
    float*  sd2   = (float*)alloc(15L * N * 4 * 4);
    float*  o2    = (float*)alloc(15L * N * 64 * 4);
    float*  E     = (float*)alloc(14L * N * 64 * 4);
    __bf16* hfR   = (__bf16*)alloc((long)N * 512 * 2);
    float*  ssR   = (float*)alloc(N * 4 * 4);
    float*  sdR   = (float*)alloc(N * 4 * 4);
    float*  Rbuf  = (float*)alloc((long)N * 128 * 4);
    float*  tmpR  = (float*)alloc((long)N * 64 * 4);
    __bf16* hfF   = (__bf16*)alloc((long)N * 512 * 2);
    float*  ssF   = (float*)alloc(N * 4 * 4);
    float*  sdF   = (float*)alloc(N * 4 * 4);
    float*  Fbuf  = (float*)alloc((long)N * 128 * 4);
    float*  tmpF  = (float*)alloc((long)N * 64 * 4);

    float* out_recon = (float*)d_out;
    float* out_fc    = out_recon + N * 64;
    float* out_E2    = out_fc + N * 64;

    auto rp = [&](const float* s, __bf16* d, int r, int c, int rs, int cs, int o) {
        repack_k<<<(r * c + 255) / 256, 256, 0, stream>>>(s, d, r, c, rs, cs, o);
    };
    rp(g1_W, g1Wb, 64, 512, 512, 1, 0);
    rp(c1_w, c1w0b, 128, 128, 2, 256, 0);   // W0^T[ci][co] = c1_w[co][ci][0]
    rp(c1_w, c1w1b, 128, 128, 2, 256, 1);
    rp(g2_W, g2Wb, 128, 256, 256, 1, 0);
    rp(c2_w, c2w0b, 64, 64, 2, 128, 0);
    rp(c2_w, c2w1b, 64, 64, 2, 128, 1);
    rp(r1_W, r1Wb, 64, 512, 512, 1, 0);
    rp(r2_W, r2Wb, 128, 64, 64, 1, 0);
    rp(r3_W, r3Wb, 64, 64, 64, 1, 0);
    rp(f1_W, f1Wb, 64, 512, 512, 1, 0);
    rp(f3_W, f3Wb, 64, 64, 64, 1, 0);

    // ---- stage 1: h1 = x @ g1_W   [16 x 1024 x 512] bf16 (packed)
    gemm_bf16_k<0><<<dim3(16, 8, 16), 128, 0, stream>>>(
        x, 1024L * 64, 64, nullptr, g1Wb, nullptr, 512, 64,
        nullptr, nullptr, hf1, 1024L * 512, 512);
    ssd_k<<<(16 * 1024 * 4 + 255) / 256, 256, 0, stream>>>(hf1, g1_as, g1_ad, ss1, sd1, 128, 16L * 1024 * 4);
    gat_attn_k<8><<<dim3(64, 16), 128, 0, stream>>>(
        hf1, 1024L * 512, ss1, sd1, 4096L, edge, 1024L * 1024, g1_b, o1, 1024L * 128, 0);
    // conv1 (k=2) + ELU: o1c[t] = elu(o1[t]@W0^T + o1[t+1]@W1^T + b)
    gemm_bf16_k<1><<<dim3(16, 2, 15), 128, 0, stream>>>(
        o1, 131072L, 128, o1 + 131072, c1w0b, c1w1b, 128, 128,
        c1_b, o1c, nullptr, 131072L, 128);

    // ---- stage 2
    gemm_bf16_k<0><<<dim3(16, 4, 15), 128, 0, stream>>>(
        o1c, 131072L, 128, nullptr, g2Wb, nullptr, 256, 128,
        nullptr, nullptr, hf2, 1024L * 256, 256);
    ssd_k<<<(15 * 1024 * 4 + 255) / 256, 256, 0, stream>>>(hf2, g2_as, g2_ad, ss2, sd2, 64, 15L * 1024 * 4);
    gat_attn_k<4><<<dim3(64, 15), 128, 0, stream>>>(
        hf2, 1024L * 256, ss2, sd2, 4096L, edge, 1024L * 1024, g2_b, o2, 65536L, 0);
    gemm_bf16_k<1><<<dim3(16, 1, 14), 128, 0, stream>>>(
        o2, 65536L, 64, o2 + 65536, c2w0b, c2w1b, 64, 64,
        c2_b, E, nullptr, 65536L, 64);

    // ---- reconstruction head (E[-1]=E[13], edge[15])
    gemm_bf16_k<0><<<dim3(16, 8, 1), 128, 0, stream>>>(
        E + 13L * 65536, 0, 64, nullptr, r1Wb, nullptr, 512, 64,
        nullptr, nullptr, hfR, 0, 512);
    ssd_k<<<(1024 * 4 + 255) / 256, 256, 0, stream>>>(hfR, r1_as, r1_ad, ssR, sdR, 128, 1024L * 4);
    gat_attn_k<8><<<dim3(64, 1), 128, 0, stream>>>(
        hfR, 0, ssR, sdR, 0, edge + 15L * 1024 * 1024, 0, r1_b, Rbuf, 0, 1);
    gemm_bf16_k<2><<<dim3(16, 1, 1), 128, 0, stream>>>(
        Rbuf, 0, 128, nullptr, r2Wb, nullptr, 64, 128, r2_b, tmpR, nullptr, 0, 64);
    gemm_bf16_k<0><<<dim3(16, 1, 1), 128, 0, stream>>>(
        tmpR, 0, 64, nullptr, r3Wb, nullptr, 64, 64, r3_b, out_recon, nullptr, 0, 64);

    // ---- forecasting head (E[-2]=E[12], edge[14])
    gemm_bf16_k<0><<<dim3(16, 8, 1), 128, 0, stream>>>(
        E + 12L * 65536, 0, 64, nullptr, f1Wb, nullptr, 512, 64,
        nullptr, nullptr, hfF, 0, 512);
    ssd_k<<<(1024 * 4 + 255) / 256, 256, 0, stream>>>(hfF, f1_as, f1_ad, ssF, sdF, 128, 1024L * 4);
    gat_attn_k<8><<<dim3(64, 1), 128, 0, stream>>>(
        hfF, 0, ssF, sdF, 0, edge + 14L * 1024 * 1024, 0, f1_b, Fbuf, 0, 1);
    gemm_bf16_k<2><<<dim3(16, 1, 1), 128, 0, stream>>>(
        Fbuf, 0, 128, nullptr, r2Wb, nullptr, 64, 128, r2_b, tmpF, nullptr, 0, 64);
    gemm_bf16_k<0><<<dim3(16, 1, 1), 128, 0, stream>>>(
        tmpF, 0, 64, nullptr, f3Wb, nullptr, 64, 64, f3_b, out_fc, nullptr, 0, 64);

    // ---- third output: E[-2]
    copy_k<<<(65536 + 255) / 256, 256, 0, stream>>>(E + 12L * 65536, out_E2, 65536);
}
